// bilstmbahdanauneer_79173427134836
// MI455X (gfx1250) — matrix operand order repaired and checked by reading, not compiled, not run on hardware
//
#include <hip/hip_runtime.h>
#include <hip/hip_bf16.h>

// ---------------------------------------------------------------------------
// Problem constants (match reference)
// ---------------------------------------------------------------------------
#define BB       4
#define TT       256
#define EE       256
#define HH       256      // HID
#define DD       512      // 2*HID
#define G4       1024     // 4*HID
#define NROW     (BB*TT)  // 1024
#define NLAB     9

typedef __attribute__((ext_vector_type(16))) __bf16 v16bf;
typedef __attribute__((ext_vector_type(8)))  __bf16 v8bf;
typedef __attribute__((ext_vector_type(8)))  float  v8f;

__device__ __forceinline__ v16bf cat8(v8bf lo, v8bf hi) {
    return __builtin_shufflevector(lo, hi, 0,1,2,3,4,5,6,7,8,9,10,11,12,13,14,15);
}

__device__ __forceinline__ v8f wmma_bf16(v16bf a, v16bf b, v8f c) {
    // D = A(16x32) * B(32x16) + C(16x16), f32 accumulate
    return __builtin_amdgcn_wmma_f32_16x16x32_bf16(false, a, false, b, (short)0, c,
                                                   false, false);
}

__device__ __forceinline__ float fast_tanh(float x) {
#if __has_builtin(__builtin_amdgcn_tanhf)
    return __builtin_amdgcn_tanhf(x);       // single V_TANH_F32 on gfx1250
#else
    return tanhf(x);
#endif
}

__device__ __forceinline__ float sigmoidf_fast(float x) {
    return 1.0f / (1.0f + __expf(-x));
}

// gfx1250 async copy: one 16-byte chunk global -> LDS, tracked by ASYNCcnt.
// vdst VGPR carries the LDS byte address (low 32 bits of generic LDS pointer).
__device__ __forceinline__ void async_ld_b128(unsigned int lds_addr, const void* gptr) {
    asm volatile("global_load_async_to_lds_b128 %0, %1, off"
                 :
                 : "v"(lds_addr), "v"((unsigned long long)(uintptr_t)gptr)
                 : "memory");
}
__device__ __forceinline__ void wait_async0() {
    asm volatile("s_wait_asynccnt 0" ::: "memory");
}

// ---------------------------------------------------------------------------
// Embedding gather -> bf16 activations, row = b*T + t
// ---------------------------------------------------------------------------
__global__ void embed_kernel(const int* __restrict__ tok,
                             const float* __restrict__ emb,
                             __bf16* __restrict__ Xbf) {
    int id  = blockIdx.x * 256 + threadIdx.x;      // NROW*EE threads
    int row = id >> 8;
    int e   = id & 255;
    int tk  = tok[row];
    Xbf[id] = (__bf16)emb[(size_t)tk * EE + e];
}

// ---------------------------------------------------------------------------
// Elementwise f32 -> bf16 cast (for activation A-matrices)
// ---------------------------------------------------------------------------
__global__ void cast_f32_bf16(const float* __restrict__ src,
                              __bf16* __restrict__ dst, int n) {
    int i = blockIdx.x * 256 + threadIdx.x;
    if (i < n) dst[i] = (__bf16)src[i];
}

// ---------------------------------------------------------------------------
// Pack a B matrix (K x N logical) into WMMA 32x16 bf16 fragment order:
//   dst[((ntile*(K/32)+kk)*32 + lane)*16 + e]
//     lanes 0-15  : n = ntile*16 + lane,       k = kk*32 + e
//     lanes 16-31 : n = ntile*16 + (lane-16),  k = kk*32 + 16 + e
// transpose==1: src is (N,K) row-major (ld=K), B[k][n] = src[n*ld + k]   (W.T)
// transpose==0: src is (K,N) row-major (ld=N), B[k][n] = src[k*ld + n]
// ---------------------------------------------------------------------------
__global__ void pack_Bfrag(const float* __restrict__ src,
                           __bf16* __restrict__ dst,
                           int K, int N, int ld, int transpose) {
    int id = blockIdx.x * 256 + threadIdx.x;
    if (id >= K * N) return;
    int e    = id & 15;
    int lane = (id >> 4) & 31;
    int kq   = K >> 5;
    int kk   = (id >> 9) % kq;
    int nt   = id / (512 * kq);
    int n    = nt * 16 + (lane & 15);
    int k    = kk * 32 + (lane >> 4) * 16 + e;
    float v  = transpose ? src[(size_t)n * ld + k] : src[(size_t)k * ld + n];
    dst[id]  = (__bf16)v;
}

// ---------------------------------------------------------------------------
// Generic bf16 WMMA GEMM: C(M,N) = A(M,K) * Bfrag(K,N) [+ bias]
// A row-major bf16 (lda), Bfrag pre-packed, C f32 (ldc).
// Block = 256 threads = 8 waves; the block's 16xK A-tile is staged into LDS
// once via gfx1250 async-to-LDS copies, then each wave computes one 16x16
// output tile, reading A fragments from LDS and streaming B from L2.
// Requires M%16==0, N%128==0, K%32==0, K<=512.
// ---------------------------------------------------------------------------
__global__ void gemm_bf16_frag(const __bf16* __restrict__ A,
                               const __bf16* __restrict__ Bfrag,
                               float* __restrict__ C,
                               const float* __restrict__ bias,
                               int K, int lda, int ldc) {
    __shared__ __align__(16) __bf16 Alds[16 * 512];   // 16 KB max

    int tid   = threadIdx.x;
    int lane  = tid & 31;
    int wave  = tid >> 5;
    int ntile = blockIdx.x * 8 + wave;
    int mtile = blockIdx.y;
    int mrow  = (lane & 15);
    int half  = lane >> 4;
    int kq    = K >> 5;

    // ---- async stage of the A tile (16 rows x K bf16) into LDS ----
    int k8 = K >> 3;                       // 16B chunks per row
    for (int ch = tid; ch < 16 * k8; ch += 256) {
        int row  = ch / k8;
        int col8 = (ch % k8) * 8;
        const __bf16* g = A + (size_t)(mtile * 16 + row) * lda + col8;
        unsigned int l  = (unsigned int)(uintptr_t)&Alds[row * K + col8];
        async_ld_b128(l, g);
    }
    wait_async0();
    __syncthreads();

    const __bf16* Abase = &Alds[mrow * K];
    const __bf16* Bbase = Bfrag + ((size_t)ntile * kq * 32 + lane) * 16;

    v8f acc = {};
    for (int kk = 0; kk < kq; ++kk) {
        int k0 = kk * 32;
        v8bf a0 = *(const v8bf*)(Abase + k0 + half * 16);
        v8bf a1 = *(const v8bf*)(Abase + k0 + half * 16 + 8);
        v8bf b0 = *(const v8bf*)(Bbase + (size_t)kk * 512);
        v8bf b1 = *(const v8bf*)(Bbase + (size_t)kk * 512 + 8);
        if (kk + 1 < kq)
            __builtin_prefetch(Bbase + (size_t)(kk + 1) * 512, 0, 3);
        acc = wmma_bf16(cat8(a0, a1), cat8(b0, b1), acc);
    }

    int col = ntile * 16 + (lane & 15);
    float bv = bias ? bias[col] : 0.0f;
#pragma unroll
    for (int r = 0; r < 8; ++r) {
        int row = mtile * 16 + half * 8 + r;
        C[(size_t)row * ldc + col] = acc[r] + bv;
    }
}

// ---------------------------------------------------------------------------
// LSTM recurrent scan: one block per direction (grid.x = 2), 512 threads.
// Per timestep: z(4,1024) = xproj[:,te,:] + h(4,256) @ Whh^T  via WMMA
// (M padded 4->16, rows 4..15 zero in LDS). xproj rows for step t+1 are
// async-prefetched into an LDS double buffer while step t's WMMA runs,
// hiding the L2 fetch behind the 512-WMMA recurrence matmul.
// ---------------------------------------------------------------------------
__global__ void lstm_scan(const float* __restrict__ xpF,
                          const float* __restrict__ xpB,
                          const __bf16* __restrict__ WhhFF,
                          const __bf16* __restrict__ WhhFB,
                          float* __restrict__ Hout) {
    __shared__ __align__(16) __bf16 hbf[16 * HH];   // A matrix, rows 4..15 = 0
    __shared__ float zbuf[BB * G4];                 // z = h @ Whh^T
    __shared__ float cbuf[BB * HH];                 // cell state
    __shared__ __align__(16) float xbuf[2][BB * G4];// double-buffered xproj rows

    int dir  = blockIdx.x;
    const float*  xp = dir ? xpB : xpF;
    const __bf16* Wf = dir ? WhhFB : WhhFF;

    int tid  = threadIdx.x;
    int lane = tid & 31;
    int wave = tid >> 5;                 // 0..15
    int mrow = lane & 15;
    int half = lane >> 4;

    for (int i = tid; i < 16 * HH; i += 512) hbf[i] = (__bf16)0.0f;
    for (int i = tid; i < BB * HH; i += 512) cbuf[i] = 0.0f;

    // prologue: fetch xproj rows for step 0
    {
        int te0 = dir ? (TT - 1) : 0;
        for (int ch = tid; ch < 1024; ch += 512) {       // 16 KB in 16B chunks
            int b  = ch >> 8;
            int c4 = (ch & 255) * 4;
            const float* g = xp + ((size_t)(b * TT + te0)) * G4 + c4;
            unsigned int l = (unsigned int)(uintptr_t)&xbuf[0][b * G4 + c4];
            async_ld_b128(l, g);
        }
    }
    wait_async0();
    __syncthreads();

    for (int t = 0; t < TT; ++t) {
        int cur = t & 1;
        int te  = dir ? (TT - 1 - t) : t;

        // ---- issue async prefetch of step t+1's xproj rows ----
        if (t + 1 < TT) {
            int ten = dir ? (TT - 2 - t) : (t + 1);
            for (int ch = tid; ch < 1024; ch += 512) {
                int b  = ch >> 8;
                int c4 = (ch & 255) * 4;
                const float* g = xp + ((size_t)(b * TT + ten)) * G4 + c4;
                unsigned int l = (unsigned int)(uintptr_t)&xbuf[cur ^ 1][b * G4 + c4];
                async_ld_b128(l, g);
            }
        }

        // ---- h @ Whh^T : 64 N-tiles, each wave owns 4 (overlaps the async) ----
#pragma unroll
        for (int i = 0; i < 4; ++i) {
            int nt = wave * 4 + i;
            const __bf16* Bbase = Wf + ((size_t)nt * 8 * 32 + lane) * 16;
            v8f acc = {};
#pragma unroll
            for (int kk = 0; kk < 8; ++kk) {
                int k0 = kk * 32;
                v8bf a0 = *(const v8bf*)&hbf[mrow * HH + k0 + half * 16];
                v8bf a1 = *(const v8bf*)&hbf[mrow * HH + k0 + half * 16 + 8];
                v8bf b0 = *(const v8bf*)(Bbase + (size_t)kk * 512);
                v8bf b1 = *(const v8bf*)(Bbase + (size_t)kk * 512 + 8);
                acc = wmma_bf16(cat8(a0, a1), cat8(b0, b1), acc);
            }
            if (half == 0) {            // rows 0..3 live in VGPRs 0..3, lanes 0..15
                int col = nt * 16 + mrow;
#pragma unroll
                for (int r = 0; r < 4; ++r) zbuf[r * G4 + col] = acc[r];
            }
        }
        __syncthreads();   // zbuf ready; matmul reads of hbf complete

        // ---- gates + state update: 4*256 items over 512 threads ----
        for (int u = tid; u < BB * HH; u += 512) {
            int b = u >> 8;
            int j = u & 255;
            const float* xr = &xbuf[cur][b * G4];
            float iv = zbuf[b * G4 + j]           + xr[j];
            float fv = zbuf[b * G4 + HH + j]      + xr[HH + j];
            float gv = zbuf[b * G4 + 2 * HH + j]  + xr[2 * HH + j];
            float ov = zbuf[b * G4 + 3 * HH + j]  + xr[3 * HH + j];
            iv = sigmoidf_fast(iv);
            fv = sigmoidf_fast(fv);
            gv = fast_tanh(gv);
            ov = sigmoidf_fast(ov);
            float c = fv * cbuf[u] + iv * gv;
            cbuf[u] = c;
            float h = ov * fast_tanh(c);
            hbf[b * HH + j] = (__bf16)h;
            Hout[((size_t)(b * TT + te)) * DD + dir * HH + j] = h;
        }

        wait_async0();     // t+1 buffer complete (was in flight during matmul)
        __syncthreads();   // publish hbf/cbuf updates + xbuf[cur^1]
    }
}

// ---------------------------------------------------------------------------
// Fused Bahdanau energy + masked softmax, one block per (b,t), 256 threads.
// e[s] = sum_d v[d]*tanh(Q[b,t,d] + K[b,s,d]); mask tok==0 -> -inf; softmax.
// Writes bf16 attention weights (A matrix for context GEMM).
// ---------------------------------------------------------------------------
__global__ void attn_energy_softmax(const float* __restrict__ Q,
                                    const float* __restrict__ Km,
                                    const float* __restrict__ vvec,
                                    const int* __restrict__ tok,
                                    __bf16* __restrict__ Wattn) {
    __shared__ float q[DD];
    __shared__ float vv[DD];
    __shared__ float red[TT];

    int b = blockIdx.x / TT;
    int t = blockIdx.x % TT;
    int tid = threadIdx.x;           // 256 = one thread per source position s

    const float* qrow = Q + ((size_t)(b * TT + t)) * DD;
    for (int i = tid; i < DD; i += 256) { q[i] = qrow[i]; vv[i] = vvec[i]; }
    __syncthreads();

    int s = tid;
    const float* krow = Km + ((size_t)(b * TT + s)) * DD;
    float e = 0.0f;
    for (int d = 0; d < DD; ++d)
        e += vv[d] * fast_tanh(q[d] + krow[d]);

    bool valid = (tok[b * TT + s] != 0);
    e = valid ? e : -__builtin_inff();

    // block max
    red[tid] = e; __syncthreads();
    for (int off = 128; off > 0; off >>= 1) {
        if (tid < off) red[tid] = fmaxf(red[tid], red[tid + off]);
        __syncthreads();
    }
    float mx = red[0]; __syncthreads();

    float ex = (valid && mx > -__builtin_inff()) ? __expf(e - mx) : 0.0f;
    red[tid] = ex; __syncthreads();
    for (int off = 128; off > 0; off >>= 1) {
        if (tid < off) red[tid] += red[tid + off];
        __syncthreads();
    }
    float sum = red[0];
    float w = (sum > 0.0f) ? (ex / sum) : 0.0f;
    Wattn[((size_t)(b * TT + t)) * TT + s] = (__bf16)w;
}

// ---------------------------------------------------------------------------
// Final classifier: out[row,l] = concat(H,ctx)[row] . Wc[l] + bc[l]
// ---------------------------------------------------------------------------
__global__ void final_kernel(const float* __restrict__ H,
                             const float* __restrict__ ctx,
                             const float* __restrict__ Wc,
                             const float* __restrict__ bc,
                             float* __restrict__ out) {
    int id = blockIdx.x * 256 + threadIdx.x;
    if (id >= NROW * NLAB) return;
    int row = id / NLAB;
    int l   = id % NLAB;
    const float* h = H   + (size_t)row * DD;
    const float* c = ctx + (size_t)row * DD;
    const float* w = Wc  + (size_t)l * (2 * DD);
    float acc = bc[l];
    for (int d = 0; d < DD; ++d)
        acc += w[d] * h[d] + w[DD + d] * c[d];
    out[id] = acc;
}

// ---------------------------------------------------------------------------
// Host orchestration
// ---------------------------------------------------------------------------
extern "C" void kernel_launch(void* const* d_in, const int* in_sizes, int n_in,
                              void* d_out, int out_size, void* d_ws, size_t ws_size,
                              hipStream_t stream) {
    const int*   tok = (const int*)d_in[0];
    const float* emb = (const float*)d_in[2];
    const float* Wih[2][2] = {{(const float*)d_in[3],  (const float*)d_in[6]},
                              {(const float*)d_in[9],  (const float*)d_in[12]}};
    const float* Whh[2][2] = {{(const float*)d_in[4],  (const float*)d_in[7]},
                              {(const float*)d_in[10], (const float*)d_in[13]}};
    const float* bg[2][2]  = {{(const float*)d_in[5],  (const float*)d_in[8]},
                              {(const float*)d_in[11], (const float*)d_in[14]}};
    const float* Wq = (const float*)d_in[15];
    const float* Wk = (const float*)d_in[16];
    const float* vv = (const float*)d_in[17];
    const float* Wc = (const float*)d_in[18];
    const float* bc = (const float*)d_in[19];
    float* out = (float*)d_out;

    // workspace carve-up
    char* p = (char*)d_ws;
    size_t off = 0;
    auto take = [&](size_t bytes) { char* q = p + off; off += (bytes + 255) & ~(size_t)255; return q; };
    __bf16* Xbf   = (__bf16*)take((size_t)NROW * DD * 2);   // layer input bf16 (max 1024x512)
    float*  H     = (float*) take((size_t)NROW * DD * 4);   // layer output / final encoder states
    float*  xpF   = (float*) take((size_t)NROW * G4 * 4);   // x-projection fwd
    float*  xpB   = (float*) take((size_t)NROW * G4 * 4);   // x-projection bwd
    __bf16* WihFF = (__bf16*)take((size_t)DD * G4 * 2);     // packed Wih^T fwd (max K=512)
    __bf16* WihFB = (__bf16*)take((size_t)DD * G4 * 2);
    __bf16* WhhFF = (__bf16*)take((size_t)HH * G4 * 2);     // packed Whh^T fwd
    __bf16* WhhFB = (__bf16*)take((size_t)HH * G4 * 2);
    float*  Qm    = (float*) take((size_t)NROW * DD * 4);
    float*  Km    = (float*) take((size_t)NROW * DD * 4);
    __bf16* WqF   = (__bf16*)take((size_t)DD * DD * 2);
    __bf16* WkF   = (__bf16*)take((size_t)DD * DD * 2);
    __bf16* Wattn = (__bf16*)take((size_t)BB * TT * TT * 2);
    __bf16* Hfrag = (__bf16*)take((size_t)BB * TT * DD * 2);
    float*  ctx   = (float*) take((size_t)NROW * DD * 4);
    __bf16* Hbf   = (__bf16*)take((size_t)NROW * DD * 2);

    // 1) embedding -> bf16 layer-0 input
    embed_kernel<<<dim3(NROW * EE / 256), dim3(256), 0, stream>>>(tok, emb, Xbf);

    // 2) LSTM layers
    for (int l = 0; l < 2; ++l) {
        int in_dim = (l == 0) ? EE : DD;

        // pack weights into WMMA fragment order (B = W^T : K=in_dim/HH, N=4H)
        int nIh = in_dim * G4;
        pack_Bfrag<<<dim3((nIh + 255) / 256), dim3(256), 0, stream>>>(Wih[l][0], WihFF, in_dim, G4, in_dim, 1);
        pack_Bfrag<<<dim3((nIh + 255) / 256), dim3(256), 0, stream>>>(Wih[l][1], WihFB, in_dim, G4, in_dim, 1);
        int nHh = HH * G4;
        pack_Bfrag<<<dim3((nHh + 255) / 256), dim3(256), 0, stream>>>(Whh[l][0], WhhFF, HH, G4, HH, 1);
        pack_Bfrag<<<dim3((nHh + 255) / 256), dim3(256), 0, stream>>>(Whh[l][1], WhhFB, HH, G4, HH, 1);

        // xproj = X @ Wih^T + b   (M=1024, N=1024, K=in_dim)
        gemm_bf16_frag<<<dim3(G4 / 128, NROW / 16), dim3(256), 0, stream>>>(
            Xbf, WihFF, xpF, bg[l][0], in_dim, in_dim, G4);
        gemm_bf16_frag<<<dim3(G4 / 128, NROW / 16), dim3(256), 0, stream>>>(
            Xbf, WihFB, xpB, bg[l][1], in_dim, in_dim, G4);

        // recurrent scan, both directions (persistent block each)
        lstm_scan<<<dim3(2), dim3(512), 0, stream>>>(xpF, xpB, WhhFF, WhhFB, H);

        // next layer input
        cast_f32_bf16<<<dim3(NROW * DD / 256), dim3(256), 0, stream>>>(H, Xbf, NROW * DD);
    }

    // 3) attention projections Q = H Wq^T, K = H Wk^T
    cast_f32_bf16<<<dim3(NROW * DD / 256), dim3(256), 0, stream>>>(H, Hbf, NROW * DD);
    pack_Bfrag<<<dim3(DD * DD / 256), dim3(256), 0, stream>>>(Wq, WqF, DD, DD, DD, 1);
    pack_Bfrag<<<dim3(DD * DD / 256), dim3(256), 0, stream>>>(Wk, WkF, DD, DD, DD, 1);
    gemm_bf16_frag<<<dim3(DD / 128, NROW / 16), dim3(256), 0, stream>>>(
        Hbf, WqF, Qm, nullptr, DD, DD, DD);
    gemm_bf16_frag<<<dim3(DD / 128, NROW / 16), dim3(256), 0, stream>>>(
        Hbf, WkF, Km, nullptr, DD, DD, DD);

    // 4) fused energy + softmax -> bf16 attention weights
    attn_energy_softmax<<<dim3(BB * TT), dim3(TT), 0, stream>>>(Qm, Km, vv, tok, Wattn);

    // 5) context = weights @ H  (per batch: M=T, N=D, K=T)
    for (int b = 0; b < BB; ++b) {
        pack_Bfrag<<<dim3(TT * DD / 256), dim3(256), 0, stream>>>(
            H + (size_t)b * TT * DD, Hfrag + (size_t)b * TT * DD, TT, DD, DD, 0);
        gemm_bf16_frag<<<dim3(DD / 128, TT / 16), dim3(256), 0, stream>>>(
            Wattn + (size_t)b * TT * TT, Hfrag + (size_t)b * TT * DD,
            ctx + (size_t)b * TT * DD, nullptr, TT, TT, DD);
    }

    // 6) classifier head
    final_kernel<<<dim3((NROW * NLAB + 255) / 256), dim3(256), 0, stream>>>(
        H, ctx, Wc, bc, out);
}
